// ST_GNN_83305185673880
// MI455X (gfx1250) — matrix operand order repaired and checked by reading, not compiled, not run on hardware
//
#include <hip/hip_runtime.h>
#include <hip/hip_bf16.h>

// ---------------------------------------------------------------------------
// ST-GNN step for MI455X (gfx1250, wave32).
// Dense GEMMs use v_wmma_f32_16x16x32_bf16 (bf16 in, f32 accum), register-
// blocked 16x64 per wave (4 WMMAs per A-load). Irregular gather/scatter uses
// global_atomic_add_f32. Force decoder fuses gather + WMMA + in-wave 64->3
// reduction with a branch-free A-gather. N=50000, E=800000: multiples of 16.
// ---------------------------------------------------------------------------

typedef __attribute__((ext_vector_type(16))) __bf16 v16bf;
typedef __attribute__((ext_vector_type(8)))  float  f32x8;

union ABf { v16bf v; unsigned int u[8]; };

#define IN_DIM 30
#define HDIM   128
#define ECH    4
#define FK_PAD 288   // 2*H + EC = 260 padded to multiple of 32

__device__ inline unsigned short f2bf(float f) {
    unsigned int u = __float_as_uint(f);
    u += 0x7FFFu + ((u >> 16) & 1u);           // round-to-nearest-even
    return (unsigned short)(u >> 16);
}
__device__ inline float sigmoidf_(float x) { return 1.0f / (1.0f + __expf(-x)); }

// A k offset used in this file (contiguous order, not AMD's table order): VGPR j holds K pair at
//   k = half*16 + 2*j
__device__ inline int a_off(int j, int half) {
    return (half << 4) + (j << 1);
}

// ---------------------------- utility kernels ------------------------------

__global__ void k_zero(float* p, long n) {
    long i = (long)blockIdx.x * blockDim.x + threadIdx.x;
    if (i < n) p[i] = 0.0f;
}

// weights: (O,K) f32 row-major -> (O,Kpad) bf16, zero padded
__global__ void k_cvt_weight(const float* __restrict__ src, unsigned short* __restrict__ dst,
                             int O, int K, int Kpad) {
    long i = (long)blockIdx.x * blockDim.x + threadIdx.x;
    if (i >= (long)O * Kpad) return;
    int o = (int)(i / Kpad), k = (int)(i % Kpad);
    dst[i] = (k < K) ? f2bf(src[(long)o * K + k]) : (unsigned short)0;
}

__global__ void k_cvt_act(const float* __restrict__ src, unsigned short* __restrict__ dst, long n) {
    long i = (long)blockIdx.x * blockDim.x + threadIdx.x;
    if (i < n) dst[i] = f2bf(src[i]);
}

// ------------------------- SAGE layer 1 (K=30, VALU) -----------------------

__global__ void k_scatter_mean30(const float* __restrict__ x, const int* __restrict__ ei,
                                 float* __restrict__ agg, float* __restrict__ cnt, int E) {
    int e = blockIdx.x * blockDim.x + threadIdx.x;
    if (e >= E) return;
    int r = ei[e], c = ei[E + e];
    const float* xs = x + (long)r * IN_DIM;
    float* d = agg + (long)c * IN_DIM;
    #pragma unroll
    for (int k = 0; k < IN_DIM; ++k) atomicAdd(d + k, xs[k]);
    atomicAdd(cnt + c, 1.0f);
}

__global__ void k_sage1(const float* __restrict__ x, const float* __restrict__ agg,
                        const float* __restrict__ cnt,
                        const float* __restrict__ W1l, const float* __restrict__ b1l,
                        const float* __restrict__ W1r,
                        float* __restrict__ x1, unsigned short* __restrict__ x1b, int N) {
    long idx = (long)blockIdx.x * blockDim.x + threadIdx.x;
    if (idx >= (long)N * HDIM) return;
    int n = (int)(idx >> 7), o = (int)(idx & 127);
    float inv = 1.0f / fmaxf(cnt[n], 1.0f);
    const float* ag = agg + (long)n * IN_DIM;
    const float* xs = x   + (long)n * IN_DIM;
    const float* wl = W1l + (long)o * IN_DIM;
    const float* wr = W1r + (long)o * IN_DIM;
    float acc = b1l[o];
    #pragma unroll
    for (int k = 0; k < IN_DIM; ++k) acc += wl[k] * (ag[k] * inv) + wr[k] * xs[k];
    acc = fmaxf(acc, 0.0f);
    x1[idx] = acc;
    x1b[idx] = f2bf(acc);
}

// ------------------------- SAGE layer 2 aggregation ------------------------

__global__ void k_scatter_sum128(const float* __restrict__ x1, const int* __restrict__ ei,
                                 float* __restrict__ agg2, int E) {
    long idx = (long)blockIdx.x * blockDim.x + threadIdx.x;
    if (idx >= (long)E * 32) return;
    int e = (int)(idx >> 5);
    int c0 = (int)((idx & 31) << 2);
    int r = ei[e], c = ei[E + e];
    float4 v = *(const float4*)(x1 + (long)r * HDIM + c0);
    float* d = agg2 + (long)c * HDIM + c0;
    atomicAdd(d + 0, v.x); atomicAdd(d + 1, v.y);
    atomicAdd(d + 2, v.z); atomicAdd(d + 3, v.w);
}

__global__ void k_agg2_finish(const float* __restrict__ agg2, const float* __restrict__ cnt,
                              unsigned short* __restrict__ agg2b, int N) {
    long idx = (long)blockIdx.x * blockDim.x + threadIdx.x;
    if (idx >= (long)N * HDIM) return;
    int n = (int)(idx >> 7);
    float inv = 1.0f / fmaxf(cnt[n], 1.0f);
    agg2b[idx] = f2bf(agg2[idx] * inv);
}

// ------------- generic single-wave register-blocked WMMA GEMM --------------
// Y[M,O] (+)= X[M,Kpad]_bf16 @ W[O,Kpad]_bf16^T  (+ bias) (opt ReLU)
// Each wave computes a 16 x (16*CT) tile: A loaded once per k-step, CT WMMAs.
// grid = (M/16, O/(16*CT)), block = 32.
// B k order used here (contiguous, same as A; not AMD's table order): lane = N col; VGPR j holds K pair
//   k = (lane>=16 ? 16:0) + 2*j
// C/D layout: VGPR r: M = r + (lane>=16 ? 8:0), N = lane&15.

#define GF_ACCUM 1
#define GF_RELU  2

template <int CT>
__global__ void k_wmma_gemm(const unsigned short* __restrict__ X,
                            const unsigned short* __restrict__ Wb,
                            const float* __restrict__ bias,
                            float* __restrict__ Y,
                            int Kpad, int O, int flags) {
    int lane = threadIdx.x;
    int half = lane >> 4, q = lane & 15;
    long rowbase = (long)blockIdx.x * 16;
    int  colbase = blockIdx.y * 16 * CT;

    const unsigned short* xrow = X + (rowbase + q) * (long)Kpad;
    const unsigned short* wrow[CT];
    #pragma unroll
    for (int t = 0; t < CT; ++t)
        wrow[t] = Wb + (long)(colbase + t * 16 + q) * Kpad;

    f32x8 c[CT];
    #pragma unroll
    for (int t = 0; t < CT; ++t) c[t] = f32x8{};
    if (flags & GF_ACCUM) {
        #pragma unroll
        for (int t = 0; t < CT; ++t)
            #pragma unroll
            for (int r = 0; r < 8; ++r)
                c[t][r] = Y[(rowbase + r + half * 8) * O + colbase + t * 16 + q];
    }

    int K32 = Kpad >> 5;
    for (int kk = 0; kk < K32; ++kk) {
        int kb = kk << 5;
        ABf a;
        #pragma unroll
        for (int j = 0; j < 8; ++j)
            a.u[j] = *(const unsigned int*)(xrow + kb + a_off(j, half));
        #pragma unroll
        for (int t = 0; t < CT; ++t) {
            ABf b;
            #pragma unroll
            for (int j = 0; j < 8; ++j)
                b.u[j] = *(const unsigned int*)(wrow[t] + kb + (half << 4) + (j << 1));
            c[t] = __builtin_amdgcn_wmma_f32_16x16x32_bf16(false, a.v, false, b.v,
                                                           (short)0, c[t], false, false);
        }
    }

    #pragma unroll
    for (int t = 0; t < CT; ++t) {
        float bv = bias ? bias[colbase + t * 16 + q] : 0.0f;
        #pragma unroll
        for (int r = 0; r < 8; ++r) {
            float v = c[t][r] + bv;
            if (flags & GF_RELU) v = fmaxf(v, 0.0f);
            Y[(rowbase + r + half * 8) * O + colbase + t * 16 + q] = v;
        }
    }
}

// ------------------------------- GRU cell ----------------------------------

__global__ void k_gru(const float* __restrict__ gi, const float* __restrict__ gh,
                      const float* __restrict__ h,
                      float* __restrict__ hnew, unsigned short* __restrict__ hnb, int N) {
    long idx = (long)blockIdx.x * blockDim.x + threadIdx.x;
    if (idx >= (long)N * HDIM) return;
    int n = (int)(idx >> 7), o = (int)(idx & 127);
    const float* gin = gi + (long)n * 384;
    const float* ghn = gh + (long)n * 384;
    float r  = sigmoidf_(gin[o]       + ghn[o]);
    float z  = sigmoidf_(gin[128 + o] + ghn[128 + o]);
    float nn = tanhf(gin[256 + o] + r * ghn[256 + o]);
    float out = (1.0f - z) * nn + z * h[idx];
    hnew[idx] = out;
    hnb[idx]  = f2bf(out);
}

// --------------------------- disp decoder tail -----------------------------

__global__ void k_disp2(const float* __restrict__ d1f, const float* __restrict__ Wd2,
                        const float* __restrict__ bd2, float* __restrict__ out, int N) {
    int n = blockIdx.x * blockDim.x + threadIdx.x;
    if (n >= N) return;
    const float* row = d1f + (long)n * 32;
    #pragma unroll
    for (int c = 0; c < 3; ++c) {
        float acc = bd2[c];
        const float* w = Wd2 + c * 32;
        #pragma unroll
        for (int k = 0; k < 32; ++k) acc += row[k] * w[k];
        out[(long)n * 3 + c] = acc;
    }
}

// --------------------------- force decoder ---------------------------------
// Fused: edge_feat = [h_new[row] | h_new[col] | edge_attr | pad] (K=288),
// one wave computes the FULL f1 tile (16 edges x 64 hidden) with 4 C blocks,
// so the random h_new gather happens once per edge. The k-loop is split into
// three BRANCH-FREE segments (h[row] blocks, h[col] blocks, edge_attr block
// built with unconditional loads + cndmask select). Then relu+bias into LDS
// and an in-wave 64->3 reduction with Wf2; direct store (no atomics).
// grid = E/16, block = 32.
__global__ void k_force(const unsigned short* __restrict__ hnb, const int* __restrict__ ei,
                        const float* __restrict__ ea,
                        const unsigned short* __restrict__ Wf1b, const float* __restrict__ bf1,
                        const float* __restrict__ Wf2, const float* __restrict__ bf2,
                        float* __restrict__ outF, int E) {
    __shared__ float tile[16][68];
    int lane = threadIdx.x;
    int half = lane >> 4, q = lane & 15;
    int e = blockIdx.x * 16 + q;

    int r0 = ei[e], r1 = ei[E + e];
    const unsigned short* h0 = hnb + (long)r0 * HDIM;
    const unsigned short* h1 = hnb + (long)r1 * HDIM;
    const float* eav = ea + (long)e * ECH;
    const unsigned short* wrow[4];
    #pragma unroll
    for (int t = 0; t < 4; ++t)
        wrow[t] = Wf1b + (long)(t * 16 + q) * FK_PAD;

    f32x8 c[4];
    #pragma unroll
    for (int t = 0; t < 4; ++t) c[t] = f32x8{};

    auto do_step = [&](const ABf& a, int kb) {
        #pragma unroll
        for (int t = 0; t < 4; ++t) {
            ABf b;
            #pragma unroll
            for (int j = 0; j < 8; ++j)
                b.u[j] = *(const unsigned int*)(wrow[t] + kb + (half << 4) + (j << 1));
            c[t] = __builtin_amdgcn_wmma_f32_16x16x32_bf16(false, a.v, false, b.v,
                                                           (short)0, c[t], false, false);
        }
    };

    // k-blocks 0..3: h_new[row]
    #pragma unroll
    for (int kk = 0; kk < 4; ++kk) {
        int kb = kk << 5;
        ABf a;
        #pragma unroll
        for (int j = 0; j < 8; ++j)
            a.u[j] = *(const unsigned int*)(h0 + kb + a_off(j, half));
        do_step(a, kb);
    }
    // k-blocks 4..7: h_new[col]
    #pragma unroll
    for (int kk = 4; kk < 8; ++kk) {
        int kb = kk << 5;
        ABf a;
        #pragma unroll
        for (int j = 0; j < 8; ++j)
            a.u[j] = *(const unsigned int*)(h1 + (kb - 128) + a_off(j, half));
        do_step(a, kb);
    }
    // k-block 8: [edge_attr(4) | zero pad] -- unconditional loads + mask select
    {
        float4 ev = *(const float4*)eav;
        unsigned int p0 = (unsigned int)f2bf(ev.x) | ((unsigned int)f2bf(ev.y) << 16);
        unsigned int p1 = (unsigned int)f2bf(ev.z) | ((unsigned int)f2bf(ev.w) << 16);
        unsigned int m = (half == 0) ? 0xFFFFFFFFu : 0u;  // k=256..259 live only in half 0
        ABf a;
        a.u[0] = p0 & m;   // k pair (256,257) = ea[0],ea[1]
        a.u[1] = p1 & m;   // k pair (258,259) = ea[2],ea[3]
        #pragma unroll
        for (int j = 2; j < 8; ++j) a.u[j] = 0u;
        do_step(a, 256);
    }

    #pragma unroll
    for (int t = 0; t < 4; ++t) {
        float bv = bf1[t * 16 + q];
        #pragma unroll
        for (int r = 0; r < 8; ++r)
            tile[r + half * 8][t * 16 + q] = fmaxf(c[t][r] + bv, 0.0f);
    }
    __syncthreads();

    if (lane < 16) {
        int eo = blockIdx.x * 16 + lane;
        float a0 = bf2[0], a1 = bf2[1], a2 = bf2[2];
        #pragma unroll
        for (int n = 0; n < 64; ++n) {
            float v = tile[lane][n];
            a0 += v * Wf2[0 * 64 + n];
            a1 += v * Wf2[1 * 64 + n];
            a2 += v * Wf2[2 * 64 + n];
        }
        outF[(long)eo * 3 + 0] = a0;
        outF[(long)eo * 3 + 1] = a1;
        outF[(long)eo * 3 + 2] = a2;
    }
}

// ------------------------------ launcher -----------------------------------

extern "C" void kernel_launch(void* const* d_in, const int* in_sizes, int n_in,
                              void* d_out, int out_size, void* d_ws, size_t ws_size,
                              hipStream_t stream) {
    const float* x   = (const float*)d_in[0];
    const int*   ei  = (const int*)  d_in[1];
    const float* ea  = (const float*)d_in[2];
    const float* h   = (const float*)d_in[3];
    const float* W1l = (const float*)d_in[4];
    const float* b1l = (const float*)d_in[5];
    const float* W1r = (const float*)d_in[6];
    const float* W2l = (const float*)d_in[7];
    const float* b2l = (const float*)d_in[8];
    const float* W2r = (const float*)d_in[9];
    const float* Wih = (const float*)d_in[10];
    const float* Whh = (const float*)d_in[11];
    const float* bih = (const float*)d_in[12];
    const float* bhh = (const float*)d_in[13];
    const float* Wd1 = (const float*)d_in[14];
    const float* bd1 = (const float*)d_in[15];
    const float* Wd2 = (const float*)d_in[16];
    const float* bd2 = (const float*)d_in[17];
    const float* Wf1 = (const float*)d_in[18];
    const float* bf1 = (const float*)d_in[19];
    const float* Wf2 = (const float*)d_in[20];
    const float* bf2 = (const float*)d_in[21];

    const int N = in_sizes[0] / IN_DIM;   // 50000
    const int E = in_sizes[1] / 2;        // 800000

    float* outDisp  = (float*)d_out;
    float* outForce = outDisp + (long)N * 3;
    float* outH     = outForce + (long)E * 3;

    // workspace layout (256B aligned slots)
    char* ws = (char*)d_ws;
    size_t off = 0;
    auto alloc = [&](size_t bytes) -> char* {
        char* p = ws + off;
        off += (bytes + 255) & ~(size_t)255;
        return p;
    };
    float*          cnt    = (float*)         alloc((size_t)N * 4);
    float*          agg1   = (float*)         alloc((size_t)N * IN_DIM * 4);
    float*          x1     = (float*)         alloc((size_t)N * HDIM * 4);
    unsigned short* x1b    = (unsigned short*)alloc((size_t)N * HDIM * 2);
    float*          agg2   = (float*)         alloc((size_t)N * HDIM * 4);
    unsigned short* agg2b  = (unsigned short*)alloc((size_t)N * HDIM * 2);
    float*          x2     = (float*)         alloc((size_t)N * HDIM * 4);
    unsigned short* x2b    = (unsigned short*)alloc((size_t)N * HDIM * 2);
    unsigned short* hb     = (unsigned short*)alloc((size_t)N * HDIM * 2);
    float*          gi     = (float*)         alloc((size_t)N * 384 * 4);
    float*          gh     = (float*)         alloc((size_t)N * 384 * 4);
    unsigned short* hnb    = (unsigned short*)alloc((size_t)N * HDIM * 2);
    float*          d1f    = (float*)         alloc((size_t)N * 32 * 4);
    unsigned short* wW2l   = (unsigned short*)alloc((size_t)HDIM * HDIM * 2);
    unsigned short* wW2r   = (unsigned short*)alloc((size_t)HDIM * HDIM * 2);
    unsigned short* wWih   = (unsigned short*)alloc((size_t)384 * HDIM * 2);
    unsigned short* wWhh   = (unsigned short*)alloc((size_t)384 * HDIM * 2);
    unsigned short* wWd1   = (unsigned short*)alloc((size_t)32 * HDIM * 2);
    unsigned short* wWf1   = (unsigned short*)alloc((size_t)64 * FK_PAD * 2);

    const int T = 256;
    auto blk = [](long n, int t) { return (unsigned)((n + t - 1) / t); };

    // 0) clear accumulators
    k_zero<<<blk((long)N, T), T, 0, stream>>>(cnt, N);
    k_zero<<<blk((long)N * IN_DIM, T), T, 0, stream>>>(agg1, (long)N * IN_DIM);
    k_zero<<<blk((long)N * HDIM, T), T, 0, stream>>>(agg2, (long)N * HDIM);

    // 1) weight + h conversion to bf16
    k_cvt_weight<<<blk((long)HDIM * HDIM, T), T, 0, stream>>>(W2l, wW2l, HDIM, HDIM, HDIM);
    k_cvt_weight<<<blk((long)HDIM * HDIM, T), T, 0, stream>>>(W2r, wW2r, HDIM, HDIM, HDIM);
    k_cvt_weight<<<blk((long)384 * HDIM, T), T, 0, stream>>>(Wih, wWih, 384, HDIM, HDIM);
    k_cvt_weight<<<blk((long)384 * HDIM, T), T, 0, stream>>>(Whh, wWhh, 384, HDIM, HDIM);
    k_cvt_weight<<<blk((long)32 * HDIM, T), T, 0, stream>>>(Wd1, wWd1, 32, HDIM, HDIM);
    k_cvt_weight<<<blk((long)64 * FK_PAD, T), T, 0, stream>>>(Wf1, wWf1, 64, 2 * HDIM + ECH, FK_PAD);
    k_cvt_act<<<blk((long)N * HDIM, T), T, 0, stream>>>(h, hb, (long)N * HDIM);

    // 2) SAGE1: scatter-mean (K=30) + node linear (VALU)
    k_scatter_mean30<<<blk(E, T), T, 0, stream>>>(x, ei, agg1, cnt, E);
    k_sage1<<<blk((long)N * HDIM, T), T, 0, stream>>>(x, agg1, cnt, W1l, b1l, W1r, x1, x1b, N);

    // 3) SAGE2 aggregation + fused WMMA linears:
    //    x2 = relu(agg2@W2l^T + b2l + x1@W2r^T)
    k_scatter_sum128<<<blk((long)E * 32, T), T, 0, stream>>>(x1, ei, agg2, E);
    k_agg2_finish<<<blk((long)N * HDIM, T), T, 0, stream>>>(agg2, cnt, agg2b, N);
    {
        dim3 g(N / 16, HDIM / 64);
        k_wmma_gemm<4><<<g, 32, 0, stream>>>(agg2b, wW2l, b2l, x2, HDIM, HDIM, 0);
        k_wmma_gemm<4><<<g, 32, 0, stream>>>(x1b,   wW2r, nullptr, x2, HDIM, HDIM, GF_ACCUM | GF_RELU);
    }
    k_cvt_act<<<blk((long)N * HDIM, T), T, 0, stream>>>(x2, x2b, (long)N * HDIM);

    // 4) GRU: gi = x2@Wih^T + bih, gh = h@Whh^T + bhh, then gates
    {
        dim3 g(N / 16, 384 / 64);
        k_wmma_gemm<4><<<g, 32, 0, stream>>>(x2b, wWih, bih, gi, HDIM, 384, 0);
        k_wmma_gemm<4><<<g, 32, 0, stream>>>(hb,  wWhh, bhh, gh, HDIM, 384, 0);
    }
    k_gru<<<blk((long)N * HDIM, T), T, 0, stream>>>(gi, gh, h, outH, hnb, N);

    // 5) disp decoder: relu(h_new@Wd1^T + bd1) @ Wd2^T + bd2
    {
        dim3 g(N / 16, 32 / 32);
        k_wmma_gemm<2><<<g, 32, 0, stream>>>(hnb, wWd1, bd1, d1f, HDIM, 32, GF_RELU);
    }
    k_disp2<<<blk(N, T), T, 0, stream>>>(d1f, Wd2, bd2, outDisp, N);

    // 6) force decoder: fused gather + full-width WMMA + in-wave reduction
    {
        dim3 g(E / 16);
        k_force<<<g, 32, 0, stream>>>(hnb, ei, ea, wWf1, bf1, Wf2, bf2, outForce, E);
    }
}
